// GIN_3100966387997
// MI455X (gfx1250) — matrix operand order repaired and checked by reading, not compiled, not run on hardware
//
#include <hip/hip_runtime.h>
#include <hip/hip_bf16.h>

#define N_NODES  50000
#define N_EDGES  800000
#define IN_CH    64
#define HID      128
#define OUT_CH   64
#define N_GRAPHS 64

typedef __attribute__((ext_vector_type(16))) _Float16 v16h;
typedef __attribute__((ext_vector_type(8)))  float    v8f;
typedef __attribute__((ext_vector_type(2)))  float    v2f;
typedef __attribute__((ext_vector_type(4)))  float    f32x4;

// ---------------- utility kernels ----------------

__global__ void copy_f32v4(const float* __restrict__ in, float* __restrict__ out, int n4) {
    int i = blockIdx.x * blockDim.x + threadIdx.x;
    if (i < n4) ((f32x4*)out)[i] = ((const f32x4*)in)[i];
}

__global__ void zero_f32(float* __restrict__ out, int n) {
    int i = blockIdx.x * blockDim.x + threadIdx.x;
    if (i < n) out[i] = 0.0f;
}

// w [K][N] f32 -> wt [N][K] f16 (transposed so B fragments read contiguous K)
__global__ void convert_wt(const float* __restrict__ w, _Float16* __restrict__ wt, int K, int N) {
    int i = blockIdx.x * blockDim.x + threadIdx.x;
    if (i < K * N) {
        int k = i / N, n = i % N;
        wt[(size_t)n * K + k] = (_Float16)w[i];
    }
}

// agg[dst[e]][c..c+3] += feat[src[e]][c..c+3]
// Wave-structured: lane -> 4-channel chunk, (wave, lane/CHUNKS) -> edge, so the
// edge index is uniform across the lanes that share it (scalarizable loads) and
// all index math is compile-time shifts. f32 hardware atomics, L2-resident dst.
template<int C>
__global__ void scatter_add(const int* __restrict__ src, const int* __restrict__ dst,
                            const float* __restrict__ feat, float* __restrict__ agg) {
    constexpr int CHUNKS = C / 4;        // 4-float chunks per edge (16 or 32)
    constexpr int EPW    = 32 / CHUNKS;  // edges per wave (2 or 1)
    constexpr int EPB    = EPW * 8;      // edges per 256-thread block
    int wave = threadIdx.x >> 5;
    int lane = threadIdx.x & 31;
    int e = blockIdx.x * EPB + wave * EPW + lane / CHUNKS;
    if (e >= N_EDGES) return;
    int c = (lane % CHUNKS) << 2;
    int s = src[e], d = dst[e];
    f32x4 v = *(const f32x4*)(feat + (size_t)s * C + c);
    float* o = agg + (size_t)d * C + c;
    unsafeAtomicAdd(o + 0, v[0]);
    unsafeAtomicAdd(o + 1, v[1]);
    unsafeAtomicAdd(o + 2, v[2]);
    unsafeAtomicAdd(o + 3, v[3]);
}

// batch is sorted: register-accumulate runs of equal graph id, one atomic per boundary.
__global__ void pool_reduce(const float* __restrict__ h, const int* __restrict__ batch,
                            float* __restrict__ pooled) {
    int c  = threadIdx.x;            // 0..127 (channel)
    int n0 = blockIdx.x * 256;
    int n1 = n0 + 256; if (n1 > N_NODES) n1 = N_NODES;
    if (n0 >= N_NODES) return;
    float acc = 0.0f;
    int cur = batch[n0];
    for (int n = n0; n < n1; ++n) {
        int g = batch[n];            // wave-uniform -> scalar load
        if (g != cur) {
            unsafeAtomicAdd(&pooled[cur * HID + c], acc);
            acc = 0.0f; cur = g;
        }
        acc += h[(size_t)n * HID + c];
    }
    unsafeAtomicAdd(&pooled[cur * HID + c], acc);
}

// ---------------- WMMA GEMM: C[M][128] = act(A[M][K] @ W[K][128] + bias) ----------------
// One wave = one 16x16 output tile; K (compile-time 64/128) fully unrolled into
// 2/4 back-to-back v_wmma_f32_16x16x32_f16 with pipelined fragment loads.
// Block = 256 threads = 8 waves covering N = 128. grid.x = M/16.
// Wt is the f16 *transposed* weight [N][K].
template<int K, bool A_HALF, bool RELU, bool OUT_HALF>
__global__ void gemm_wmma(const void* __restrict__ Av, const _Float16* __restrict__ Wt,
                          const float* __restrict__ bias,
                          float* __restrict__ Cf, _Float16* __restrict__ Ch) {
    constexpr int N = HID;
    int lane = threadIdx.x & 31;
    int wave = threadIdx.x >> 5;
    int r16  = lane & 15;
    int half = lane >> 4;
    int mBase = blockIdx.x << 4;
    int nBase = wave << 4;
    int row = mBase + r16;

    v8f acc = {};
#pragma unroll
    for (int kb = 0; kb < K; kb += 32) {
        v16h a, b;
        // A fragment (16x32 f16): element e -> k = kb + 16*(e/8) + 8*half + e%8
        if (A_HALF) {
            const _Float16* p = (const _Float16*)Av + (size_t)row * K + kb + half * 8;
            ((f32x4*)&a)[0] = *(const f32x4*)(p);        // 8 halfs: k .. k+7
            ((f32x4*)&a)[1] = *(const f32x4*)(p + 16);   // 8 halfs: k+16 .. k+23
        } else {
            const float* Af = (const float*)Av + (size_t)row * K + kb + half * 8;
            f32x4 x0 = *(const f32x4*)(Af);
            f32x4 x1 = *(const f32x4*)(Af + 4);
            f32x4 x2 = *(const f32x4*)(Af + 16);
            f32x4 x3 = *(const f32x4*)(Af + 20);
#pragma unroll
            for (int e = 0; e < 4; ++e) {
                a[e]      = (_Float16)x0[e];
                a[4 + e]  = (_Float16)x1[e];
                a[8 + e]  = (_Float16)x2[e];
                a[12 + e] = (_Float16)x3[e];
            }
        }
        // B fragment (32x16 f16): lane holds column nBase+r16, k = kb + 16*(e/8) + 8*half + e%8
        const _Float16* bp = Wt + (size_t)(nBase + r16) * K + kb + half * 8;
        ((f32x4*)&b)[0] = *(const f32x4*)(bp);
        ((f32x4*)&b)[1] = *(const f32x4*)(bp + 16);

        acc = __builtin_amdgcn_wmma_f32_16x16x32_f16(false, a, false, b,
                                                     (short)0, acc, false, false);
    }

    int col = nBase + r16;
    float bv = bias[col];
#pragma unroll
    for (int r = 0; r < 8; ++r) {
        float v = acc[r] + bv;
        if (RELU) v = fmaxf(v, 0.0f);
        int m = mBase + half * 8 + r;
        if (OUT_HALF) Ch[(size_t)m * N + col] = (_Float16)v;
        else          Cf[(size_t)m * N + col] = v;
    }
}

// ---------------- final 64x128 @ 128x64 in full f32 (v_wmma_f32_16x16x4_f32) ----------------
// pooled sums are large (up to ~1e5); keep f32. One block, 16 waves = 4x4 tiles.
__global__ void final_gemm(const float* __restrict__ pooled, const float* __restrict__ wl,
                           const float* __restrict__ bl, float* __restrict__ out) {
    int lane = threadIdx.x & 31;
    int wave = threadIdx.x >> 5;     // 0..15
    int r16  = lane & 15;
    int half = lane >> 4;
    int mBase = (wave >> 2) << 4;
    int nBase = (wave & 3) << 4;
    int row = mBase + r16;
    int col = nBase + r16;

    v8f acc = {};
#pragma unroll
    for (int kb = 0; kb < HID; kb += 4) {
        int ka = kb + half * 2;
        v2f a, b;
        // A 16x4 f32: VGPR0 = K=2*half, VGPR1 = K=2*half+1 (contiguous pair)
        v2f ap = *(const v2f*)(pooled + row * HID + ka);
        a[0] = ap[0]; a[1] = ap[1];
        // B 4x16 f32: symmetric layout, column per lane
        b[0] = wl[(ka    ) * OUT_CH + col];
        b[1] = wl[(ka + 1) * OUT_CH + col];
        acc = __builtin_amdgcn_wmma_f32_16x16x4_f32(false, a, false, b,
                                                    (short)0, acc, false, false);
    }
    float bv = bl[col];
#pragma unroll
    for (int r = 0; r < 8; ++r)
        out[(mBase + half * 8 + r) * OUT_CH + col] = acc[r] + bv;
}

// ---------------- host launch ----------------

extern "C" void kernel_launch(void* const* d_in, const int* in_sizes, int n_in,
                              void* d_out, int out_size, void* d_ws, size_t ws_size,
                              hipStream_t stream) {
    const float* x   = (const float*)d_in[0];
    const int*   ei  = (const int*)d_in[1];
    const int*   bat = (const int*)d_in[2];
    const float* w1a = (const float*)d_in[3];
    const float* b1a = (const float*)d_in[4];
    const float* w1b = (const float*)d_in[5];
    const float* b1b = (const float*)d_in[6];
    const float* w2a = (const float*)d_in[7];
    const float* b2a = (const float*)d_in[8];
    const float* w2b = (const float*)d_in[9];
    const float* b2b = (const float*)d_in[10];
    const float* wl  = (const float*)d_in[11];
    const float* bl  = (const float*)d_in[12];
    const int* src = ei;
    const int* dst = ei + N_EDGES;
    float* out = (float*)d_out;

    char* ws = (char*)d_ws;
    float*    agg    = (float*)   (ws + 0);          // 25,600,000 B (agg1 then agg2)
    _Float16* hHalf  = (_Float16*)(ws + 25600000);   // 12,800,000 B (h1_f16 then h2_f16)
    float*    hF32   = (float*)   (ws + 38400000);   // 25,600,000 B (h1b then h2b)
    _Float16* w1aT   = (_Float16*)(ws + 64000000);   // 16,384 B
    _Float16* w1bT   = (_Float16*)(ws + 64016384);   // 32,768 B
    _Float16* w2aT   = (_Float16*)(ws + 64049152);   // 32,768 B
    _Float16* w2bT   = (_Float16*)(ws + 64081920);   // 32,768 B
    float*    pooled = (float*)   (ws + 64114688);   // 32,768 B

    // weight prep (every call — deterministic, tiny)
    convert_wt<<<(IN_CH * HID + 255) / 256, 256, 0, stream>>>(w1a, w1aT, IN_CH, HID);
    convert_wt<<<(HID * HID + 255) / 256, 256, 0, stream>>>(w1b, w1bT, HID, HID);
    convert_wt<<<(HID * HID + 255) / 256, 256, 0, stream>>>(w2a, w2aT, HID, HID);
    convert_wt<<<(HID * HID + 255) / 256, 256, 0, stream>>>(w2b, w2bT, HID, HID);

    // ---- layer 1: agg = x + sum_{src->dst} x[src] ----
    copy_f32v4<<<(N_NODES * IN_CH / 4 + 255) / 256, 256, 0, stream>>>(x, agg, N_NODES * IN_CH / 4);
    scatter_add<IN_CH><<<(N_EDGES + 15) / 16, 256, 0, stream>>>(src, dst, x, agg);
    // h1 = relu(agg @ w1a + b1a) -> f16
    gemm_wmma<IN_CH, false, true, true><<<N_NODES / 16, 256, 0, stream>>>(agg, w1aT, b1a, nullptr, hHalf);
    // h1b = relu(h1 @ w1b + b1b) -> f32   (outer relu of layer 1 fused)
    gemm_wmma<HID, true, true, false><<<N_NODES / 16, 256, 0, stream>>>(hHalf, w1bT, b1b, hF32, nullptr);

    // ---- layer 2 ----
    copy_f32v4<<<(N_NODES * HID / 4 + 255) / 256, 256, 0, stream>>>(hF32, agg, N_NODES * HID / 4);
    scatter_add<HID><<<(N_EDGES + 7) / 8, 256, 0, stream>>>(src, dst, hF32, agg);
    gemm_wmma<HID, false, true, true><<<N_NODES / 16, 256, 0, stream>>>(agg, w2aT, b2a, nullptr, hHalf);
    gemm_wmma<HID, true, true, false><<<N_NODES / 16, 256, 0, stream>>>(hHalf, w2bT, b2b, hF32, nullptr);

    // ---- pooling over sorted batch ids ----
    zero_f32<<<(N_GRAPHS * HID + 255) / 256, 256, 0, stream>>>(pooled, N_GRAPHS * HID);
    pool_reduce<<<(N_NODES + 255) / 256, 128, 0, stream>>>(hF32, bat, pooled);

    // ---- final linear, full f32 WMMA ----
    final_gemm<<<1, 512, 0, stream>>>(pooled, wl, bl, out);
}
